// MultiHeadAttention_15762529976429
// MI455X (gfx1250) — matrix operand order repaired and checked by reading, not compiled, not run on hardware
//
#include <hip/hip_runtime.h>

#define BS 4
#define SEQ 2048
#define DM 1024
#define NH 16
#define DH 64

typedef __bf16 bf16;
typedef __attribute__((ext_vector_type(16))) __bf16 bf16x16;
typedef __attribute__((ext_vector_type(8)))  float  f32x8;
typedef __attribute__((ext_vector_type(4)))  unsigned int u32x4;

#if __has_builtin(__builtin_amdgcn_global_load_async_to_lds_b128)
#define USE_ASYNC_LDS 1
// exact param types per hipcc diagnostics:
//   arg0: int __attribute__((vector_size(16))) __device__ *  (global src, AS1)
//   arg1: int __attribute__((vector_size(16))) __shared__ *  (LDS dst,   AS3)
typedef int v4i __attribute__((vector_size(16)));
typedef __attribute__((address_space(1))) v4i* as1vp;
typedef __attribute__((address_space(3))) v4i* as3vp;
#endif

__device__ __forceinline__ void wait_async0() {
#if defined(USE_ASYNC_LDS)
#if __has_builtin(__builtin_amdgcn_s_wait_asynccnt)
  __builtin_amdgcn_s_wait_asynccnt(0);
#else
  asm volatile("s_wait_asynccnt 0" ::: "memory");
#endif
#endif
}

union FragU { bf16x16 v; u32x4 q[2]; };

// A-matrix fragment (16x32): lane half lh: K chunks [lh*8, lh*8+8) and [16+lh*8, ...)
__device__ __forceinline__ bf16x16 load_afrag(const bf16* rowk0, int lh) {
  FragU f;
  f.q[0] = *reinterpret_cast<const u32x4*>(rowk0 + lh * 8);
  f.q[1] = *reinterpret_cast<const u32x4*>(rowk0 + 16 + lh * 8);
  return f.v;
}

// B-matrix fragment (32x16): lane half lh: K chunks [lh*8, lh*8+8) and [16+lh*8, ...)
__device__ __forceinline__ bf16x16 load_bfrag(const bf16* rowk0, int lh) {
  FragU f;
  f.q[0] = *reinterpret_cast<const u32x4*>(rowk0 + lh * 8);
  f.q[1] = *reinterpret_cast<const u32x4*>(rowk0 + 16 + lh * 8);
  return f.v;
}

__device__ __forceinline__ f32x8 wmma_bf16(bf16x16 a, bf16x16 b, f32x8 c) {
  return __builtin_amdgcn_wmma_f32_16x16x32_bf16(false, a, false, b, (short)0, c,
                                                 false, false);
}

// ---------------------------------------------------------------- convert
__global__ __launch_bounds__(256) void cvt_f32_bf16(const float* __restrict__ in,
                                                    bf16* __restrict__ out, int n) {
  int i = (blockIdx.x * blockDim.x + threadIdx.x) * 4;
  if (i + 3 < n) {
    float4 v = *reinterpret_cast<const float4*>(in + i);
    out[i + 0] = (bf16)v.x;
    out[i + 1] = (bf16)v.y;
    out[i + 2] = (bf16)v.z;
    out[i + 3] = (bf16)v.w;
  }
}

// ---------------------------------------------------------------- GEMM: C = A @ W^T + bias
// Block tile 64x128, 4 waves, wave tile 32x64 (8 WMMA / 6 frag-loads per K step).
// mode 0: store bf16 head-split [B,H,S,64]   mode 1: store fp32 flat [M,1024]
__global__ __launch_bounds__(128) void gemm_wmma(const bf16* __restrict__ A,
                                                 const bf16* __restrict__ W,
                                                 const float* __restrict__ bias,
                                                 void* __restrict__ out, int mode) {
  const int m0 = blockIdx.x * 64;
  const int n0 = blockIdx.y * 128;
  const int tid = threadIdx.x;
  const int w = tid >> 5;
  const int wm = (w >> 1) * 32;
  const int wn = (w & 1) * 64;
  const int lane = tid & 31;
  const int lrow = lane & 15;
  const int lh = lane >> 4;

  f32x8 acc[2][4] = {};

  const bf16* ar[2];
  const bf16* wr[4];
#pragma unroll
  for (int i = 0; i < 2; ++i) ar[i] = A + (size_t)(m0 + wm + i * 16 + lrow) * DM;
#pragma unroll
  for (int j = 0; j < 4; ++j) wr[j] = W + (size_t)(n0 + wn + j * 16 + lrow) * DM;

  for (int k0 = 0; k0 < DM; k0 += 32) {
    bf16x16 af[2], bfm[4];
#pragma unroll
    for (int i = 0; i < 2; ++i) af[i] = load_afrag(ar[i] + k0, lh);
#pragma unroll
    for (int j = 0; j < 4; ++j) bfm[j] = load_bfrag(wr[j] + k0, lh);
    __builtin_prefetch(ar[0] + k0 + 256, 0, 1);
    __builtin_prefetch(wr[0] + k0 + 256, 0, 1);
#pragma unroll
    for (int i = 0; i < 2; ++i)
#pragma unroll
      for (int j = 0; j < 4; ++j) acc[i][j] = wmma_bf16(af[i], bfm[j], acc[i][j]);
  }

  const int lh8 = lh * 8;
#pragma unroll
  for (int i = 0; i < 2; ++i)
#pragma unroll
    for (int j = 0; j < 4; ++j) {
      const int ncol = n0 + wn + j * 16 + lrow;
      const float bv = bias[ncol];
#pragma unroll
      for (int r = 0; r < 8; ++r) {
        const int mrow = m0 + wm + i * 16 + r + lh8;
        const float v = acc[i][j][r] + bv;
        if (mode == 0) {
          const int b = mrow >> 11, s = mrow & (SEQ - 1);
          const int h = ncol >> 6, dh = ncol & (DH - 1);
          ((bf16*)out)[(((size_t)(b * NH + h) * SEQ) + s) * DH + dh] = (bf16)v;
        } else {
          ((float*)out)[(size_t)mrow * DM + ncol] = v;
        }
      }
    }
}

// ---------------------------------------------------------------- flash attention
// q,k,v: [B,H,S,64] bf16.  ctx out: [B,S,1024] bf16 (head-merged).
__global__ __launch_bounds__(128) void flash_attn(const bf16* __restrict__ q,
                                                  const bf16* __restrict__ k,
                                                  const bf16* __restrict__ v,
                                                  const int* __restrict__ mask,
                                                  bf16* __restrict__ ctx) {
  __shared__ __align__(16) bf16 lds_k[32 * DH];     // K tile, row-major [key][dh]
  __shared__ __align__(16) bf16 lds_vt[DH * 32];    // V^T tile [dh][key]
  __shared__ __align__(16) bf16 lds_p[4 * 16 * 32]; // per-wave P tile 16x32

  const int qb = blockIdx.x * 64;
  const int h = blockIdx.y;
  const int b = blockIdx.z;
  const int tid = threadIdx.x;
  const int w = tid >> 5;
  const int lane = tid & 31;
  const int lrow = lane & 15;
  const int lh = lane >> 4;

  const size_t bh = (size_t)(b * NH + h) * SEQ;

  // Q fragments held in registers for the whole kernel (16 rows x 64 K)
  const bf16* qp = q + (bh + qb + w * 16 + lrow) * DH;
  const bf16x16 qa0 = load_afrag(qp, lh);
  const bf16x16 qa1 = load_afrag(qp + 32, lh);

  f32x8 cacc[4] = {};
  float mrow[8], lsum[8];
#pragma unroll
  for (int r = 0; r < 8; ++r) { mrow[r] = -3.0e38f; lsum[r] = 0.0f; }

  const int* maskp = mask + b * SEQ;

  for (int kb = 0; kb < SEQ; kb += 32) {
    __syncthreads();  // protect LDS reuse from previous iteration

    // ---- K tile: 32x64 row-major, staged once per block (async if available)
    {
      const bf16* gp = k + (bh + kb) * DH + tid * 16;
      bf16* lp = &lds_k[tid * 16];
#if defined(USE_ASYNC_LDS)
      __builtin_amdgcn_global_load_async_to_lds_b128((as1vp)(void*)gp,
                                                     (as3vp)(void*)lp, 0, 0);
      __builtin_amdgcn_global_load_async_to_lds_b128((as1vp)(void*)(gp + 8),
                                                     (as3vp)(void*)(lp + 8), 0, 0);
#else
      *reinterpret_cast<u32x4*>(lp) = *reinterpret_cast<const u32x4*>(gp);
      *reinterpret_cast<u32x4*>(lp + 8) = *reinterpret_cast<const u32x4*>(gp + 8);
#endif
    }
    // ---- V^T staging: lds_vt[dh][kk] = V[kb+kk][dh]
    {
      const int r = tid >> 2;        // key row 0..31
      const int nb = (tid & 3) * 16; // dh block
      const bf16* vp = v + (bh + kb + r) * DH + nb;
      FragU f;
      f.q[0] = *reinterpret_cast<const u32x4*>(vp);
      f.q[1] = *reinterpret_cast<const u32x4*>(vp + 8);
#pragma unroll
      for (int i = 0; i < 16; ++i) lds_vt[(nb + i) * 32 + r] = f.v[i];
    }
    wait_async0();
    __syncthreads();

    // ---- S = q . k^T for 32 keys (two 16-col C tiles), K frags from LDS
    f32x8 s[2];
#pragma unroll
    for (int ct = 0; ct < 2; ++ct) {
      const bf16* kp = &lds_k[(ct * 16 + lrow) * DH];
      bf16x16 kf0 = load_bfrag(kp, lh);
      bf16x16 kf1 = load_bfrag(kp + 32, lh);
      f32x8 c = {};
      c = wmma_bf16(qa0, kf0, c);
      c = wmma_bf16(qa1, kf1, c);
      const int col = kb + ct * 16 + lrow;
      const bool dead = (maskp[col] == 0);
#pragma unroll
      for (int r = 0; r < 8; ++r) {
        float sv = c[r] * 0.125f; // 1/sqrt(64)
        c[r] = dead ? -1.0e9f : sv;
      }
      s[ct] = c;
    }

    // ---- online softmax (row reductions across the 16-lane half)
#pragma unroll
    for (int r = 0; r < 8; ++r) {
      float mx = fmaxf(s[0][r], s[1][r]);
#pragma unroll
      for (int off = 8; off >= 1; off >>= 1) mx = fmaxf(mx, __shfl_xor(mx, off, 16));
      const float mnew = fmaxf(mrow[r], mx);
      const float p0 = __expf(s[0][r] - mnew);
      const float p1 = __expf(s[1][r] - mnew);
      float rs = p0 + p1;
#pragma unroll
      for (int off = 8; off >= 1; off >>= 1) rs += __shfl_xor(rs, off, 16);
      const float scale = __expf(mrow[r] - mnew);
      lsum[r] = lsum[r] * scale + rs;
      mrow[r] = mnew;
#pragma unroll
      for (int d = 0; d < 4; ++d) cacc[d][r] *= scale;
      const int prow = r + lh * 8;
      lds_p[w * 512 + prow * 32 + lrow] = (bf16)p0;
      lds_p[w * 512 + prow * 32 + 16 + lrow] = (bf16)p1;
    }
    asm volatile("s_wait_dscnt 0" ::: "memory"); // wave-private LDS RAW

    // ---- ctx += P (16x32) @ V (32x64)
    bf16x16 pa = load_afrag(&lds_p[w * 512 + lrow * 32], lh);
#pragma unroll
    for (int d = 0; d < 4; ++d) {
      bf16x16 vf = load_bfrag(&lds_vt[(d * 16 + lrow) * 32], lh);
      cacc[d] = wmma_bf16(pa, vf, cacc[d]);
    }
  }

  // ---- normalize + store ctx (merged heads: [b, s, h*64+dh])
#pragma unroll
  for (int r = 0; r < 8; ++r) {
    const int row = qb + w * 16 + r + lh * 8;
    const float inv = 1.0f / lsum[r];
#pragma unroll
    for (int d = 0; d < 4; ++d) {
      const int col = h * DH + d * 16 + lrow;
      ctx[((size_t)(b * SEQ) + row) * DM + col] = (bf16)(cacc[d][r] * inv);
    }
  }
}

// ---------------------------------------------------------------- launcher
extern "C" void kernel_launch(void* const* d_in, const int* in_sizes, int n_in,
                              void* d_out, int out_size, void* d_ws, size_t ws_size,
                              hipStream_t stream) {
  const float* Q  = (const float*)d_in[0];
  const float* K  = (const float*)d_in[1];
  const float* V  = (const float*)d_in[2];
  const int*   mk = (const int*)d_in[3];
  const float* Wq = (const float*)d_in[4];
  const float* bq = (const float*)d_in[5];
  const float* Wk = (const float*)d_in[6];
  const float* bk = (const float*)d_in[7];
  const float* Wv = (const float*)d_in[8];
  const float* bv = (const float*)d_in[9];
  const float* Wo = (const float*)d_in[10];
  const float* bo = (const float*)d_in[11];

  const size_t MTOT = (size_t)BS * SEQ; // 8192
  const size_t NACT = MTOT * DM;        // 8.39M
  const size_t NW   = (size_t)DM * DM;  // 1.05M

  char* ws = (char*)d_ws;
  size_t off = 0;
  auto alloc = [&](size_t bytes) -> void* {
    void* p = ws + off;
    off += (bytes + 255) & ~(size_t)255;
    return p;
  };
  bf16* Qb  = (bf16*)alloc(NACT * 2);
  bf16* Kb  = (bf16*)alloc(NACT * 2);
  bf16* Vb  = (bf16*)alloc(NACT * 2);
  bf16* Wqb = (bf16*)alloc(NW * 2);
  bf16* Wkb = (bf16*)alloc(NW * 2);
  bf16* Wvb = (bf16*)alloc(NW * 2);
  bf16* Wob = (bf16*)alloc(NW * 2);
  bf16* qh  = (bf16*)alloc(NACT * 2); // [B,H,S,64]
  bf16* kh  = (bf16*)alloc(NACT * 2);
  bf16* vh  = (bf16*)alloc(NACT * 2);
  bf16* ctx = (bf16*)alloc(NACT * 2); // [B,S,1024]

  {
    int ga = (int)(NACT / 4 / 256);
    int gw = (int)(NW / 4 / 256);
    cvt_f32_bf16<<<ga, 256, 0, stream>>>(Q, Qb, (int)NACT);
    cvt_f32_bf16<<<ga, 256, 0, stream>>>(K, Kb, (int)NACT);
    cvt_f32_bf16<<<ga, 256, 0, stream>>>(V, Vb, (int)NACT);
    cvt_f32_bf16<<<gw, 256, 0, stream>>>(Wq, Wqb, (int)NW);
    cvt_f32_bf16<<<gw, 256, 0, stream>>>(Wk, Wkb, (int)NW);
    cvt_f32_bf16<<<gw, 256, 0, stream>>>(Wv, Wvb, (int)NW);
    cvt_f32_bf16<<<gw, 256, 0, stream>>>(Wo, Wob, (int)NW);
  }

  // projections with fused bias + head-split
  dim3 gg(MTOT / 64, DM / 128);
  gemm_wmma<<<gg, 128, 0, stream>>>(Qb, Wqb, bq, qh, 0);
  gemm_wmma<<<gg, 128, 0, stream>>>(Kb, Wkb, bk, kh, 0);
  gemm_wmma<<<gg, 128, 0, stream>>>(Vb, Wvb, bv, vh, 0);

  // attention
  dim3 ga(SEQ / 64, NH, BS);
  flash_attn<<<ga, 128, 0, stream>>>(qh, kh, vh, mk, ctx);

  // output projection -> fp32 d_out
  gemm_wmma<<<gg, 128, 0, stream>>>(ctx, Wob, bo, d_out, 1);
}
